// LSTMModel_90993177133693
// MI455X (gfx1250) — compile-verified, matrix operand order checked by reading, not run on hardware
//
#include <hip/hip_runtime.h>
#include <hip/hip_bf16.h>

// ---------------------------------------------------------------------------
// Types for CDNA5 WMMA (wave32): v_wmma_f32_16x16x32_bf16
// ---------------------------------------------------------------------------
typedef __attribute__((ext_vector_type(16))) __bf16 bf16x16;
typedef __attribute__((ext_vector_type(8)))  __bf16 bf16x8;
typedef __attribute__((ext_vector_type(8)))  float  f32x8;
typedef __attribute__((ext_vector_type(4)))  unsigned int u32x4;
typedef __attribute__((ext_vector_type(8)))  int    i32x8;
typedef __attribute__((ext_vector_type(4)))  int    i32x4;

union Frag16 {
    bf16x16 v;
    struct { bf16x8 lo; bf16x8 hi; } p;
};

__device__ __forceinline__ f32x8 wmma_bf16(bf16x16 a, bf16x16 b, f32x8 c) {
    // (neg_a, A, neg_b, B, c_mod, C, reuse_a, reuse_b)
    return __builtin_amdgcn_wmma_f32_16x16x32_bf16(false, a, false, b,
                                                   (short)0, c, false, false);
}

// Load a 16x32 bf16 A/B fragment from a row-major matrix with leading dim ld.
// Per the CDNA5 ISA 16-bit A-matrix layout:
//   lanes 0-15  hold K = {k0+0..7,  k0+16..23}
//   lanes 16-31 hold K = {k0+8..15, k0+24..31}
// Two 128-bit loads per lane.
__device__ __forceinline__ bf16x16 load_frag(const __bf16* __restrict__ base,
                                             int ld, int row16, int k0,
                                             int lane) {
    const int r  = row16 + (lane & 15);
    const int kh = (lane >> 4) << 3;              // 0 or 8
    const __bf16* p = base + (size_t)r * ld + k0 + kh;
    Frag16 f;
    f.p.lo = *(const bf16x8*)(p);
    f.p.hi = *(const bf16x8*)(p + 16);
    return f.v;
}

// ---------------------------------------------------------------------------
// Model dimensions (fixed by the reference)
// ---------------------------------------------------------------------------
#define MB   32          // batch
#define MT   512         // sequence length
#define ME   1024        // embedding dim (== layer-1 input dim 2H)
#define MH   512         // hidden
#define M4H  2048        // 4*H (gate dim)
#define MROWS (MT * MB)  // 16384 rows for the batched input GEMM
#define RG   16          // workgroups per direction in the recurrence
#define RJ   32          // H-columns per recurrence workgroup

// ---------------------------------------------------------------------------
// Tensor Data Mover: 1-D global->LDS copy of 32 KB (h_prev staging).
// Descriptor per CDNA5 ISA:
//  group0: count=1 | lds_addr | global_addr | type=2
//  group1: data_size=2 (4B) | tensor_dim0=8192 | tensor_dim1=1 |
//          tile_dim0=8192 | tile_dim1=1 | tensor_dim0_stride=8192
//  groups 2/3: zero (<=2-D tensor)
// ---------------------------------------------------------------------------
#if defined(__gfx1250__) && __has_builtin(__builtin_amdgcn_tensor_load_to_lds)
#define HAVE_TDM 1
#endif

__device__ __forceinline__ void tdm_load_32k(unsigned lds_addr,
                                             unsigned long long gaddr) {
#if defined(HAVE_TDM)
    u32x4 g0;
    g0[0] = 1u;                                        // count = 1
    g0[1] = lds_addr;                                  // LDS byte address
    g0[2] = (unsigned)(gaddr & 0xffffffffu);           // global_addr[31:0]
    g0[3] = (unsigned)((gaddr >> 32) & 0x01ffffffu)    // global_addr[56:32]
            | 0x80000000u;                             // type = 2 ("image")
    i32x8 g1;
    g1[0] = (int)(2u << 16);        // data_size = 2 -> 4-byte elements
    g1[1] = (int)(0x2000u << 16);   // tensor_dim0 = 8192 (low 16 in [63:48])
    g1[2] = (int)0x00010000;        // tensor_dim0 hi=0, tensor_dim1 = 1
    g1[3] = (int)0x20000000;        // tile_dim0 = 8192 ([127:112])
    g1[4] = 1;                      // tile_dim1 = 1, tile_dim2 = 0
    g1[5] = 8192;                   // tensor_dim0_stride = 8192
    g1[6] = 0;
    g1[7] = 0;
    i32x4 z4 = {0, 0, 0, 0};
#if __clang_major__ >= 23
    i32x8 z8 = {0, 0, 0, 0, 0, 0, 0, 0};
    __builtin_amdgcn_tensor_load_to_lds(g0, g1, z4, z4, z8, 0);
#else
    __builtin_amdgcn_tensor_load_to_lds(g0, g1, z4, z4, 0);
#endif
    __builtin_amdgcn_s_wait_tensorcnt(0);
#else
    (void)lds_addr; (void)gaddr;
#endif
}

// ---------------------------------------------------------------------------
// Embedding gather + fp32 -> bf16, output [T,B,E] row-major
// ---------------------------------------------------------------------------
__global__ void embed_kernel(const int* __restrict__ inp,
                             const float* __restrict__ emb,
                             __bf16* __restrict__ x0) {
    size_t idx = (size_t)blockIdx.x * blockDim.x + threadIdx.x;
    const size_t total = (size_t)MROWS * ME;
    if (idx >= total) return;
    int e  = (int)(idx % ME);
    size_t tb = idx / ME;          // row = t*B + b
    int b = (int)(tb % MB);
    int t = (int)(tb / MB);
    int tok = inp[b * MT + t];     // inp is [B,T]
    x0[idx] = (__bf16)emb[(size_t)tok * ME + e];
}

// fp32 -> bf16 elementwise
__global__ void cvt_bf16_kernel(const float* __restrict__ src,
                                __bf16* __restrict__ dst, size_t n) {
    size_t i = (size_t)blockIdx.x * blockDim.x + threadIdx.x;
    if (i < n) dst[i] = (__bf16)src[i];
}

// bias = bih + bhh
__global__ void bias_kernel(const float* __restrict__ bih,
                            const float* __restrict__ bhh,
                            float* __restrict__ bc, int n) {
    int i = blockIdx.x * blockDim.x + threadIdx.x;
    if (i < n) bc[i] = bih[i] + bhh[i];
}

// ---------------------------------------------------------------------------
// Batched input projection: XP[dir][m][g] = sum_k A[m][k] * W[dir][g][k] + bias[dir][g]
//   A:[MROWS,K] bf16, W:[2][M4H,K] bf16, bias:[2][M4H] f32, XP:[2][MROWS,M4H] bf16
// Block = 256 threads = 8 waves arranged 4(M) x 2(N); wave tile = 16M x 64N.
// Block tile = 64M x 128N. grid = (MROWS/64, M4H/128, 2 dirs).
// ---------------------------------------------------------------------------
__global__ __launch_bounds__(256)
void gemm_xp_kernel(const __bf16* __restrict__ A,
                    const __bf16* __restrict__ W,
                    const float* __restrict__ bias,
                    __bf16* __restrict__ XP, int K) {
    const int dir = blockIdx.z;
    const __bf16* Wd = W + (size_t)dir * M4H * K;
    const float*  bd = bias + dir * M4H;
    __bf16* xp = XP + (size_t)dir * MROWS * M4H;

    const int w    = threadIdx.x >> 5;
    const int lane = threadIdx.x & 31;
    const int wm = w >> 1, wn = w & 1;
    const int m0 = blockIdx.x * 64 + wm * 16;
    const int n0 = blockIdx.y * 128 + wn * 64;

    f32x8 acc[4];
#pragma unroll
    for (int a = 0; a < 4; ++a) acc[a] = (f32x8)(0.0f);

    for (int k = 0; k < K; k += 32) {
        // prefetch next K panel of A (speculative, GL2)
        if (k + 256 < K)
            __builtin_prefetch(A + (size_t)(m0 + (lane & 15)) * K + k + 256, 0, 1);
        bf16x16 af = load_frag(A, K, m0, k, lane);
#pragma unroll
        for (int a = 0; a < 4; ++a) {
            bf16x16 bf = load_frag(Wd, K, n0 + a * 16, k, lane);
            acc[a] = wmma_bf16(af, bf, acc[a]);
        }
    }
    // C layout: lane 0-15 col = lane, VGPR r -> row m0+r; lanes 16-31 row +8
    const int l15 = lane & 15, mh = (lane >> 4) << 3;
#pragma unroll
    for (int a = 0; a < 4; ++a) {
        const int n = n0 + a * 16 + l15;
        const float bv = bd[n];
#pragma unroll
        for (int r = 0; r < 8; ++r) {
            const int m = m0 + r + mh;
            xp[(size_t)m * M4H + n] = (__bf16)(acc[a][r] + bv);
        }
    }
}

// ---------------------------------------------------------------------------
// Persistent bidirectional LSTM recurrence for one layer.
// grid = (RG, 2 dirs), block = 256 (8 waves).
// WG g owns h-columns [g*32, g*32+32) => 128 gate rows of Whh.
// Per step: TDM-stage h_prev (32KB) into LDS; wave w computes N-tile w (16 of
// 128 gate cols) for both 16-row M tiles via 16 K-steps of bf16 WMMA (A from
// LDS); gates -> LDS; fused activations + cell update (c in LDS); h double-
// buffered in global; device-scope monotonic atomic barrier across 16 WGs/dir.
// ---------------------------------------------------------------------------
__global__ __launch_bounds__(256, 1)
void lstm_recur_kernel(const __bf16* __restrict__ XP,   // [2][T*B][4H]
                       const __bf16* __restrict__ Whh,  // [2][4H][H]
                       __bf16* __restrict__ hbuf,       // [2][2][B][H]
                       __bf16* __restrict__ OUT,        // [T*B][1024]
                       unsigned* __restrict__ ctr) {    // [2] padded by 16
    const int dir = blockIdx.y;
    const int g   = blockIdx.x;
    const int j0  = g * RJ;
    const __bf16* xp = XP + (size_t)dir * MROWS * M4H;
    const __bf16* Wd = Whh + (size_t)dir * M4H * MH;
    __bf16* hb = hbuf + (size_t)dir * 2 * MB * MH;
    unsigned* cnt = ctr + dir * 16;

    __shared__ __align__(16) __bf16 sh_h[MB * MH];   // 32 KB staged h_prev
    __shared__ float sh_gates[MB][128];              // 16 KB gate pre-acts
    __shared__ float sh_c[MB][RJ];                   //  4 KB cell state

    const int tid  = threadIdx.x;
    const int w    = tid >> 5;
    const int lane = tid & 31;
    const int l15  = lane & 31 & 15;
    const int mh   = (lane >> 4) << 3;

    // LDS byte address of sh_h (low 32 bits of the generic pointer)
    const unsigned sh_h_lds = (unsigned)(unsigned long long)(void*)&sh_h[0];

    for (int i = tid; i < MB * RJ; i += 256) sh_c[i >> 5][i & 31] = 0.0f;

    for (int step = 0; step < MT; ++step) {
        const int tt = dir ? (MT - 1 - step) : step;
        const __bf16* hprev = hb + (size_t)(step & 1) * MB * MH;
        __bf16* hnext = hb + (size_t)((step + 1) & 1) * MB * MH;

        __syncthreads();   // previous step's LDS consumers done
#if defined(HAVE_TDM)
        if (w == 0) {      // one wave issues the TDM copy, waits on TENSORcnt
            tdm_load_32k(sh_h_lds, (unsigned long long)(const void*)hprev);
        }
#else
        for (int i = tid; i < (MB * MH) / 8; i += 256)
            *(bf16x8*)(sh_h + i * 8) = *(const bf16x8*)(hprev + i * 8);
#endif
        __syncthreads();

        // ---- gates = h_prev @ Whh_slice^T  (M=32, N=128, K=512) ----
        {
            const int ncol = w * 16 + l15;                 // 0..127
            const int gate = ncol >> 5, jj = ncol & 31;
            const __bf16* brow = Wd + (size_t)(gate * MH + j0 + jj) * MH;
            const __bf16* a0row = sh_h + (size_t)l15 * MH;         // rows 0-15
            const __bf16* a1row = sh_h + (size_t)(16 + l15) * MH;  // rows 16-31
            const int kh = (lane >> 4) << 3;
            f32x8 acc0 = (f32x8)(0.0f), acc1 = (f32x8)(0.0f);
            for (int k = 0; k < MH; k += 32) {
                Frag16 bf;
                bf.p.lo = *(const bf16x8*)(brow + k + kh);
                bf.p.hi = *(const bf16x8*)(brow + k + 16 + kh);
                Frag16 a0;
                a0.p.lo = *(const bf16x8*)(a0row + k + kh);
                a0.p.hi = *(const bf16x8*)(a0row + k + 16 + kh);
                acc0 = wmma_bf16(a0.v, bf.v, acc0);
                Frag16 a1;
                a1.p.lo = *(const bf16x8*)(a1row + k + kh);
                a1.p.hi = *(const bf16x8*)(a1row + k + 16 + kh);
                acc1 = wmma_bf16(a1.v, bf.v, acc1);
            }
#pragma unroll
            for (int r = 0; r < 8; ++r) {
                sh_gates[r + mh][ncol]      = acc0[r];
                sh_gates[16 + r + mh][ncol] = acc1[r];
            }
        }
        __syncthreads();

        // ---- fused activations + cell/hidden update ----
        for (int i = tid; i < MB * RJ; i += 256) {
            const int b = i >> 5, jj = i & 31;
            const __bf16* xr = xp + ((size_t)tt * MB + b) * M4H;
            float xi = sh_gates[b][jj]      + (float)xr[0 * MH + j0 + jj];
            float xf = sh_gates[b][32 + jj] + (float)xr[1 * MH + j0 + jj];
            float xg = sh_gates[b][64 + jj] + (float)xr[2 * MH + j0 + jj];
            float xo = sh_gates[b][96 + jj] + (float)xr[3 * MH + j0 + jj];
            float iv = 1.0f / (1.0f + __expf(-xi));
            float fv = 1.0f / (1.0f + __expf(-xf));
            float gv = tanhf(xg);
            float ov = 1.0f / (1.0f + __expf(-xo));
            float cv = fv * sh_c[b][jj] + iv * gv;
            sh_c[b][jj] = cv;
            float hv = ov * tanhf(cv);
            __bf16 hb16 = (__bf16)hv;
            hnext[b * MH + j0 + jj] = hb16;
            OUT[((size_t)tt * MB + b) * (2 * MH) + dir * MH + j0 + jj] = hb16;
        }

        // ---- device-scope barrier across the 16 WGs of this direction ----
        __threadfence();
        __syncthreads();
        if (tid == 0) {
            __hip_atomic_fetch_add(cnt, 1u, __ATOMIC_RELEASE,
                                   __HIP_MEMORY_SCOPE_AGENT);
            const unsigned target = (unsigned)(RG * (step + 1));
            while (__hip_atomic_load(cnt, __ATOMIC_ACQUIRE,
                                     __HIP_MEMORY_SCOPE_AGENT) < target)
                __builtin_amdgcn_s_sleep(1);
        }
        __syncthreads();
    }
}

// ---------------------------------------------------------------------------
// Final pooled linear: out[b] = dot(pooled[b], Wout) + bout
//   pooled[b] = concat( h_f[T-1,b, :H], h_r[0,b, H:] ) from xfin [T*B][2H] bf16
// ---------------------------------------------------------------------------
__global__ __launch_bounds__(256)
void final_kernel(const __bf16* __restrict__ xfin,
                  const float* __restrict__ Wout,
                  const float* __restrict__ bout,
                  float* __restrict__ out) {
    const int b = blockIdx.x;
    __shared__ float red[256];
    const __bf16* fr = xfin + ((size_t)(MT - 1) * MB + b) * (2 * MH);
    const __bf16* rr = xfin + (size_t)b * (2 * MH);
    float s = 0.0f;
    for (int j = threadIdx.x; j < MH; j += 256) {
        s += (float)fr[j] * Wout[j];
        s += (float)rr[MH + j] * Wout[MH + j];
    }
    red[threadIdx.x] = s;
    __syncthreads();
    for (int o = 128; o > 0; o >>= 1) {
        if (threadIdx.x < o) red[threadIdx.x] += red[threadIdx.x + o];
        __syncthreads();
    }
    if (threadIdx.x == 0) out[b] = red[0] + bout[0];
}

// ---------------------------------------------------------------------------
// Host orchestration
// ---------------------------------------------------------------------------
extern "C" void kernel_launch(void* const* d_in, const int* in_sizes, int n_in,
                              void* d_out, int out_size, void* d_ws,
                              size_t ws_size, hipStream_t stream) {
    const int*   inp  = (const int*)d_in[0];
    const float* emb  = (const float*)d_in[1];
    const float* Wih  = (const float*)d_in[2];   // [2][2][4H][E]
    const float* Whh  = (const float*)d_in[3];   // [2][2][4H][H]
    const float* bih  = (const float*)d_in[4];   // [2][2][4H]
    const float* bhh  = (const float*)d_in[5];
    const float* Wout = (const float*)d_in[6];   // [1][2H]
    const float* bout = (const float*)d_in[7];   // [1]
    float* out = (float*)d_out;

    // ---- workspace carve-out ----
    char* ws = (char*)d_ws;
    size_t off = 0;
    auto carve = [&](size_t bytes) -> void* {
        void* p = ws + off;
        off = (off + bytes + 255) & ~(size_t)255;
        return p;
    };
    __bf16* x0    = (__bf16*)carve((size_t)MROWS * ME * 2);          // 32 MB
    __bf16* x1    = (__bf16*)carve((size_t)MROWS * (2 * MH) * 2);    // 32 MB
    __bf16* xfin  = (__bf16*)carve((size_t)MROWS * (2 * MH) * 2);    // 32 MB
    __bf16* xp    = (__bf16*)carve((size_t)2 * MROWS * M4H * 2);     // 128 MB
    __bf16* WihB  = (__bf16*)carve((size_t)2 * 2 * M4H * ME * 2);    // 16 MB
    __bf16* WhhB  = (__bf16*)carve((size_t)2 * 2 * M4H * MH * 2);    // 8 MB
    float*  biasC = (float*)carve((size_t)2 * 2 * M4H * 4);
    __bf16* hbuf  = (__bf16*)carve((size_t)2 * 2 * MB * MH * 2);
    unsigned* ctrs = (unsigned*)carve(2 * 16 * sizeof(unsigned));

    // ---- conversions ----
    {
        size_t n = (size_t)2 * 2 * M4H * ME;   // Wih
        cvt_bf16_kernel<<<dim3((unsigned)((n + 255) / 256)), 256, 0, stream>>>(
            Wih, WihB, n);
        n = (size_t)2 * 2 * M4H * MH;          // Whh
        cvt_bf16_kernel<<<dim3((unsigned)((n + 255) / 256)), 256, 0, stream>>>(
            Whh, WhhB, n);
        int nb = 2 * 2 * M4H;
        bias_kernel<<<(nb + 255) / 256, 256, 0, stream>>>(bih, bhh, biasC, nb);
    }

    // ---- embedding gather ----
    {
        size_t n = (size_t)MROWS * ME;
        embed_kernel<<<dim3((unsigned)((n + 255) / 256)), 256, 0, stream>>>(
            inp, emb, x0);
    }

    // ---- two layers ----
    for (int l = 0; l < 2; ++l) {
        const __bf16* Ain = (l == 0) ? x0 : x1;
        __bf16* hsOut = (l == 0) ? x1 : xfin;
        const __bf16* Wl = WihB + (size_t)l * 2 * M4H * ME;
        const __bf16* Rl = WhhB + (size_t)l * 2 * M4H * MH;
        const float*  bl = biasC + (size_t)l * 2 * M4H;

        dim3 ggrid(MROWS / 64, M4H / 128, 2);
        gemm_xp_kernel<<<ggrid, 256, 0, stream>>>(Ain, Wl, bl, xp, ME);

        hipMemsetAsync(hbuf, 0, (size_t)2 * 2 * MB * MH * 2, stream);
        hipMemsetAsync(ctrs, 0, 2 * 16 * sizeof(unsigned), stream);

        dim3 rgrid(RG, 2);
        lstm_recur_kernel<<<rgrid, 256, 0, stream>>>(xp, Rl, hbuf, hsOut, ctrs);
    }

    // ---- final linear ----
    final_kernel<<<MB, 256, 0, stream>>>(xfin, Wout, bout, out);

    (void)in_sizes; (void)n_in; (void)out_size; (void)ws_size;
}
